// LSTMEncDec_25623774888595
// MI455X (gfx1250) — matrix operand order repaired and checked by reading, not compiled, not run on hardware
//
#include <hip/hip_runtime.h>
#include <hip/hip_bf16.h>
#include <math.h>

// ---------------- types ----------------
typedef __attribute__((ext_vector_type(16))) __bf16 v16bf;
typedef __attribute__((ext_vector_type(8)))  float  v8f;

struct U4 { unsigned int a, b, c, d; };
struct U8 { U4 lo, hi; };

__device__ __forceinline__ unsigned short f2bfu(float x) {
  unsigned u = __builtin_bit_cast(unsigned, x);
  return (unsigned short)((u + 0x7FFFu + ((u >> 16) & 1u)) >> 16);
}
__device__ __forceinline__ float sigf(float x) { return 1.0f / (1.0f + expf(-x)); }

// ---------------- WMMA fragment loaders (bf16 in memory, row-major [rows, Kpad]) ----------------
// A 16x32: lanes 0-15 row=lane, K 0..15; lanes 16-31 row=lane-16, K 16..31.
// Per lane: two contiguous 16B chunks at k+16*kh and k+16*kh+8.
__device__ __forceinline__ v16bf load_a_bf(const unsigned short* __restrict__ A, int kpad,
                                           int m, int kblock, int kh) {
  const unsigned short* p = A + (size_t)m * kpad + kblock + 16 * kh;
  U8 u;
  u.lo = *(const U4*)(p);
  u.hi = *(const U4*)(p + 8);
  return __builtin_bit_cast(v16bf, u);
}

// B 32x16: lanes 0-15 K 0..15, lanes 16-31 K 16..31, N = lane&15. One contiguous 32B chunk.
__device__ __forceinline__ v16bf load_b_bf(const unsigned short* __restrict__ W, int kpad,
                                           int n, int kblock, int kh) {
  const unsigned short* p = W + (size_t)n * kpad + kblock + 16 * kh;
  U8 u;
  u.lo = *(const U4*)(p);
  u.hi = *(const U4*)(p + 8);
  return __builtin_bit_cast(v16bf, u);
}

// ---------------- dual-A WMMA GEMM: C = A1@W1^T (+ A2@W2^T) (+ bias) (+ addend) ----------------
// One wave per 16x32 output tile (two adjacent 16x16 N-tiles share the A fragment).
__global__ void __launch_bounds__(256)
gemm_dual_wmma(const unsigned short* __restrict__ A1, int kpa1, int K1,
               const unsigned short* __restrict__ W1, int kpw1,
               const unsigned short* __restrict__ A2, int kpa2, int K2,
               const unsigned short* __restrict__ W2, int kpw2,
               const float* __restrict__ bias,
               const float* __restrict__ addend, int ldadd,
               float* __restrict__ C, int ldc, int N32tiles, int totalTiles) {
  int wave = threadIdx.x >> 5;
  int lane = threadIdx.x & 31;
  int tile = blockIdx.x * (blockDim.x >> 5) + wave;
  if (tile >= totalTiles) return;            // wave-uniform: EXEC all-1s inside
  int mt = tile / N32tiles, nt = tile % N32tiles;
  int kh = lane >> 4, r = lane & 15;
  int m_a = mt * 16 + r;
  int nb0 = nt * 32 + r;
  int nb1 = nb0 + 16;

  v8f acc0 = {0.f, 0.f, 0.f, 0.f, 0.f, 0.f, 0.f, 0.f};
  v8f acc1 = {0.f, 0.f, 0.f, 0.f, 0.f, 0.f, 0.f, 0.f};
  for (int k = 0; k < K1; k += 32) {
    v16bf a  = load_a_bf(A1, kpa1, m_a, k, kh);
    v16bf b0 = load_b_bf(W1, kpw1, nb0, k, kh);
    v16bf b1 = load_b_bf(W1, kpw1, nb1, k, kh);
    acc0 = __builtin_amdgcn_wmma_f32_16x16x32_bf16(false, a, false, b0, (short)0, acc0, false, false);
    acc1 = __builtin_amdgcn_wmma_f32_16x16x32_bf16(false, a, false, b1, (short)0, acc1, false, false);
  }
  if (A2) {
    for (int k = 0; k < K2; k += 32) {
      v16bf a  = load_a_bf(A2, kpa2, m_a, k, kh);
      v16bf b0 = load_b_bf(W2, kpw2, nb0, k, kh);
      v16bf b1 = load_b_bf(W2, kpw2, nb1, k, kh);
      acc0 = __builtin_amdgcn_wmma_f32_16x16x32_bf16(false, a, false, b0, (short)0, acc0, false, false);
      acc1 = __builtin_amdgcn_wmma_f32_16x16x32_bf16(false, a, false, b1, (short)0, acc1, false, false);
    }
  }
  // C/D layout: N = lane&15 (both halves), M = v + 8*kh
  float bv0 = bias ? bias[nb0] : 0.f;
  float bv1 = bias ? bias[nb1] : 0.f;
#pragma unroll
  for (int v = 0; v < 8; ++v) {
    int m_out = mt * 16 + v + 8 * kh;
    float v0 = acc0[v] + bv0;
    float v1 = acc1[v] + bv1;
    if (addend) {
      v0 += addend[(size_t)m_out * ldadd + nb0];
      v1 += addend[(size_t)m_out * ldadd + nb1];
    }
    C[(size_t)m_out * ldc + nb0] = v0;
    C[(size_t)m_out * ldc + nb1] = v1;
  }
}

// ---------------- helpers ----------------
__global__ void cvt_w_bf16(const float* __restrict__ src, unsigned short* __restrict__ dst,
                           int N, int K, int Kpad) {
  int idx = blockIdx.x * blockDim.x + threadIdx.x;
  if (idx >= N * Kpad) return;
  int k = idx % Kpad, n = idx / Kpad;
  float v = (k < K) ? src[(size_t)n * K + k] : 0.f;
  dst[idx] = f2bfu(v);
}

__global__ void embed_pad_bf16(const int* __restrict__ questions, const float* __restrict__ embed_w,
                               unsigned short* __restrict__ X, int L, int NC, int S, int E, int Ep) {
  int idx = blockIdx.x * blockDim.x + threadIdx.x;
  if (idx >= L * S * Ep) return;
  int k = idx % Ep; int rest = idx / Ep;
  int s = rest % S; int t = rest / S;
  int b = s / NC, q = s % NC;
  int tok = questions[(b * NC + q) * L + t];
  X[idx] = (k < E) ? f2bfu(embed_w[(size_t)tok * E + k]) : (unsigned short)0;
}

__global__ void zero_f(float* __restrict__ p, int n) {
  int i = blockIdx.x * blockDim.x + threadIdx.x;
  if (i < n) p[i] = 0.f;
}

__global__ void bcast_init(const float* __restrict__ vh1, const float* __restrict__ vc1,
                           const float* __restrict__ vh2, const float* __restrict__ vc2,
                           float* __restrict__ th1, float* __restrict__ tc1,
                           float* __restrict__ th2, float* __restrict__ tc2,
                           unsigned short* __restrict__ th1bf, unsigned short* __restrict__ th2bf,
                           int S, int NC, int H) {
  int idx = blockIdx.x * blockDim.x + threadIdx.x;
  if (idx >= S * H) return;
  int s = idx / H, j = idx % H;
  size_t src = (size_t)(s / NC) * H + j;
  float a = vh1[src], b = vh2[src];
  th1[idx] = a; tc1[idx] = vc1[src];
  th2[idx] = b; tc2[idx] = vc2[src];
  th1bf[idx] = f2bfu(a); th2bf[idx] = f2bfu(b);
}

// PyTorch LSTMCell gate order i,f,g,o. Optional ragged mask (t < qlen[b]).
// Emits fp32 masked state + bf16 mirrors (masked state / raw) for downstream WMMA GEMMs.
__global__ void lstm_cell(const float* __restrict__ gates,
                          const float* __restrict__ c_prev, const float* __restrict__ h_prev,
                          float* __restrict__ h_state, float* __restrict__ c_state,
                          unsigned short* __restrict__ h_state_bf,
                          unsigned short* __restrict__ h_raw_bf,
                          const int* __restrict__ qlen, int t, int NC, int M, int H) {
  int idx = blockIdx.x * blockDim.x + threadIdx.x;
  if (idx >= M * H) return;
  int m = idx / H, j = idx % H;
  const float* g = gates + (size_t)m * 4 * H;
  float gi = sigf(g[j]);
  float gf = sigf(g[H + j]);
  float gg = tanhf(g[2 * H + j]);
  float go = sigf(g[3 * H + j]);
  float cp = c_prev[idx];
  float c = gf * cp + gi * gg;
  float h = go * tanhf(c);
  if (h_raw_bf) h_raw_bf[idx] = f2bfu(h);
  bool keep = true;
  if (qlen) keep = (t < qlen[m / NC]);
  float hs = keep ? h : h_prev[idx];
  h_state[idx] = hs;
  c_state[idx] = keep ? c : cp;
  if (h_state_bf) h_state_bf[idx] = f2bfu(hs);
}

__global__ void concat2_bf16(const float* __restrict__ a, const float* __restrict__ b,
                             unsigned short* __restrict__ out, int S, int H) {
  int idx = blockIdx.x * blockDim.x + threadIdx.x;
  if (idx >= S * 2 * H) return;
  int s = idx / (2 * H), j = idx % (2 * H);
  out[idx] = f2bfu((j < H) ? a[(size_t)s * H + j] : b[(size_t)s * H + j - H]);
}

// one wave (32 lanes) per row: out[s] = fe[s,:] . w2 + b2
__global__ void final_dot(const float* __restrict__ fe, const float* __restrict__ w2,
                          const float* __restrict__ b2, float* __restrict__ out, int K) {
  int s = blockIdx.x;
  int lane = threadIdx.x;
  float acc = 0.f;
  for (int k = lane; k < K; k += 32) acc += fe[(size_t)s * K + k] * w2[k];
  for (int off = 16; off > 0; off >>= 1) acc += __shfl_xor(acc, off, 32);
  if (lane == 0) out[s] = acc + b2[0];
}

__global__ void argmax_k(const float* __restrict__ outs, float* __restrict__ pred, int B, int NC) {
  int b = blockIdx.x * blockDim.x + threadIdx.x;
  if (b >= B) return;
  int best = 0; float bv = outs[(size_t)b * NC];
  for (int q = 1; q < NC; ++q) {
    float v = outs[(size_t)b * NC + q];
    if (v > bv) { bv = v; best = q; }
  }
  pred[b] = (float)best;
}

// ---------------- launch ----------------
extern "C" void kernel_launch(void* const* d_in, const int* in_sizes, int n_in,
                              void* d_out, int out_size, void* d_ws, size_t ws_size,
                              hipStream_t stream) {
  (void)in_sizes; (void)n_in; (void)out_size; (void)ws_size;
  constexpr int B = 64, NC = 5, L = 20, F = 20, C = 8192, E = 300, Ep = 320, H = 512;
  constexpr int G4 = 4 * H;            // 2048
  constexpr int S = B * NC;            // 320

  const float* vf      = (const float*)d_in[0];
  const int*   questions = (const int*)d_in[1];
  const int*   qlen    = (const int*)d_in[2];
  const float* embed_w = (const float*)d_in[3];
  const float* Wih_t1  = (const float*)d_in[4];
  const float* Whh_t1  = (const float*)d_in[5];
  const float* b_t1    = (const float*)d_in[6];
  const float* Wih_t2  = (const float*)d_in[7];
  const float* Whh_t2  = (const float*)d_in[8];
  const float* b_t2    = (const float*)d_in[9];
  const float* Wih_v1  = (const float*)d_in[10];
  const float* Whh_v1  = (const float*)d_in[11];
  const float* b_v1    = (const float*)d_in[12];
  const float* Wih_v2  = (const float*)d_in[13];
  const float* Whh_v2  = (const float*)d_in[14];
  const float* b_v2    = (const float*)d_in[15];
  const float* dec1_w  = (const float*)d_in[16];
  const float* dec1_b  = (const float*)d_in[17];
  const float* dec2_w  = (const float*)d_in[18];
  const float* dec2_b  = (const float*)d_in[19];
  float* out_f = (float*)d_out;

  // workspace carve-up (256B aligned)
  char* base = (char*)d_ws;
  size_t off = 0;
  auto alloc = [&](size_t bytes) { size_t o = off; off += (bytes + 255) & ~(size_t)255; return o; };
  // bf16 weights [N, Kpad]
  unsigned short* bWv1ih = (unsigned short*)(base + alloc((size_t)G4 * C * 2));
  unsigned short* bWv1hh = (unsigned short*)(base + alloc((size_t)G4 * H * 2));
  unsigned short* bWv2ih = (unsigned short*)(base + alloc((size_t)G4 * H * 2));
  unsigned short* bWv2hh = (unsigned short*)(base + alloc((size_t)G4 * H * 2));
  unsigned short* bWt1ih = (unsigned short*)(base + alloc((size_t)G4 * Ep * 2));
  unsigned short* bWt1hh = (unsigned short*)(base + alloc((size_t)G4 * H * 2));
  unsigned short* bWt2ih = (unsigned short*)(base + alloc((size_t)G4 * H * 2));
  unsigned short* bWt2hh = (unsigned short*)(base + alloc((size_t)G4 * H * 2));
  unsigned short* bWdec1 = (unsigned short*)(base + alloc((size_t)(2 * H) * (2 * H) * 2));
  // bf16 activations
  unsigned short* vfbf   = (unsigned short*)(base + alloc((size_t)B * F * C * 2));
  unsigned short* Xtext  = (unsigned short*)(base + alloc((size_t)L * S * Ep * 2));
  unsigned short* vh1bf  = (unsigned short*)(base + alloc((size_t)B * H * 2));   // contiguous pair
  unsigned short* vh2bf  = (unsigned short*)(base + alloc((size_t)B * H * 2));
  unsigned short* th1bf  = (unsigned short*)(base + alloc((size_t)S * H * 2));
  unsigned short* th2bf  = (unsigned short*)(base + alloc((size_t)S * H * 2));
  unsigned short* th1rbf = (unsigned short*)(base + alloc((size_t)S * H * 2));
  unsigned short* sTbf   = (unsigned short*)(base + alloc((size_t)S * 2 * H * 2));
  // fp32 buffers
  float* Gv1    = (float*)(base + alloc((size_t)B * F * G4 * 4));   // [b][t][4H]
  float* gatesV = (float*)(base + alloc((size_t)B * G4 * 4));
  float* gatesT = (float*)(base + alloc((size_t)S * G4 * 4));
  float* vh1 = (float*)(base + alloc((size_t)B * H * 4));           // 4 contiguous
  float* vc1 = (float*)(base + alloc((size_t)B * H * 4));
  float* vh2 = (float*)(base + alloc((size_t)B * H * 4));
  float* vc2 = (float*)(base + alloc((size_t)B * H * 4));
  float* th1 = (float*)(base + alloc((size_t)S * H * 4));
  float* tc1 = (float*)(base + alloc((size_t)S * H * 4));
  float* th2 = (float*)(base + alloc((size_t)S * H * 4));
  float* tc2 = (float*)(base + alloc((size_t)S * H * 4));
  float* fe  = (float*)(base + alloc((size_t)S * 2 * H * 4));

  auto cvt = [&](const float* src, unsigned short* dst, int N, int K, int Kpad) {
    int n = N * Kpad;
    cvt_w_bf16<<<(n + 255) / 256, 256, 0, stream>>>(src, dst, N, K, Kpad);
  };
  // 1) weight + activation conversions fp32 -> bf16
  cvt(Wih_v1, bWv1ih, G4, C, C);
  cvt(Whh_v1, bWv1hh, G4, H, H);
  cvt(Wih_v2, bWv2ih, G4, H, H);
  cvt(Whh_v2, bWv2hh, G4, H, H);
  cvt(Wih_t1, bWt1ih, G4, E, Ep);
  cvt(Whh_t1, bWt1hh, G4, H, H);
  cvt(Wih_t2, bWt2ih, G4, H, H);
  cvt(Whh_t2, bWt2hh, G4, H, H);
  cvt(dec1_w, bWdec1, 2 * H, 2 * H, 2 * H);
  cvt(vf, vfbf, B * F, C, C);  // video features, rows = (b,t)

  // 2) embedding gather + zero-pad to Ep (bf16)
  {
    int n = L * S * Ep;
    embed_pad_bf16<<<(n + 255) / 256, 256, 0, stream>>>(questions, embed_w, Xtext, L, NC, S, E, Ep);
  }
  // 3) zero video states: fp32 (4*B*H floats) and bf16 mirrors (2*B*H ushort = B*H floats)
  zero_f<<<(4 * B * H + 255) / 256, 256, 0, stream>>>(vh1, 4 * B * H);
  zero_f<<<(B * H + 255) / 256, 256, 0, stream>>>((float*)vh1bf, B * H);

  auto gemm = [&](const unsigned short* A1, int kpa1, int K1, const unsigned short* W1, int kpw1,
                  const unsigned short* A2, int kpa2, int K2, const unsigned short* W2, int kpw2,
                  const float* bias, const float* addend, int ldadd,
                  float* Cout, int ldc, int M, int N) {
    int N32 = N / 32;
    int total = (M / 16) * N32;
    gemm_dual_wmma<<<(total + 7) / 8, 256, 0, stream>>>(
        A1, kpa1, K1, W1, kpw1, A2, kpa2, K2, W2, kpw2, bias, addend, ldadd, Cout, ldc, N32, total);
  };

  // 4) hoisted video input projection: Gv1[b][t] = vf[b,t,:]@Wih_v1^T + b_v1  (M=1280,N=2048,K=8192)
  gemm(vfbf, C, C, bWv1ih, C, nullptr, 0, 0, nullptr, 0, b_v1, nullptr, 0, Gv1, G4, B * F, G4);

  // 5) video recurrence
  for (int t = 0; t < F; ++t) {
    gemm(vh1bf, H, H, bWv1hh, H, nullptr, 0, 0, nullptr, 0,
         nullptr, Gv1 + (size_t)t * G4, F * G4, gatesV, G4, B, G4);
    lstm_cell<<<(B * H + 255) / 256, 256, 0, stream>>>(gatesV, vc1, vh1, vh1, vc1, vh1bf, nullptr,
                                                       nullptr, t, NC, B, H);
    gemm(vh1bf, H, H, bWv2ih, H, vh2bf, H, H, bWv2hh, H, b_v2, nullptr, 0, gatesV, G4, B, G4);
    lstm_cell<<<(B * H + 255) / 256, 256, 0, stream>>>(gatesV, vc2, vh2, vh2, vc2, vh2bf, nullptr,
                                                       nullptr, t, NC, B, H);
  }

  // 6) seed text states from final video states (broadcast over NC)
  bcast_init<<<(S * H + 255) / 256, 256, 0, stream>>>(vh1, vc1, vh2, vc2, th1, tc1, th2, tc2,
                                                      th1bf, th2bf, S, NC, H);

  // 7) text recurrence (ragged mask)
  for (int t = 0; t < L; ++t) {
    gemm(Xtext + (size_t)t * S * Ep, Ep, Ep, bWt1ih, Ep, th1bf, H, H, bWt1hh, H,
         b_t1, nullptr, 0, gatesT, G4, S, G4);
    lstm_cell<<<(S * H + 255) / 256, 256, 0, stream>>>(gatesT, tc1, th1, th1, tc1, th1bf, th1rbf,
                                                       qlen, t, NC, S, H);
    gemm(th1rbf, H, H, bWt2ih, H, th2bf, H, H, bWt2hh, H, b_t2, nullptr, 0, gatesT, G4, S, G4);
    lstm_cell<<<(S * H + 255) / 256, 256, 0, stream>>>(gatesT, tc2, th2, th2, tc2, th2bf, nullptr,
                                                       qlen, t, NC, S, H);
  }

  // 8) head: sT = [h1 ; h2] (bf16) ; fe = sT@dec1^T + dec1_b ; out = fe@dec2^T + dec2_b
  concat2_bf16<<<(S * 2 * H + 255) / 256, 256, 0, stream>>>(th1, th2, sTbf, S, H);
  gemm(sTbf, 2 * H, 2 * H, bWdec1, 2 * H, nullptr, 0, 0, nullptr, 0,
       dec1_b, nullptr, 0, fe, 2 * H, S, 2 * H);
  final_dot<<<S, 32, 0, stream>>>(fe, dec2_w, dec2_b, out_f, 2 * H);
  // 9) predictions (stored as float indices after the 320 outputs)
  argmax_k<<<1, 64, 0, stream>>>(out_f, out_f + S, B, NC);
}
